// FS2_One_Shot_RANet_9294309228576
// MI455X (gfx1250) — matrix operand order repaired and checked by reading, not compiled, not run on hardware
//
#include <hip/hip_runtime.h>
#include <math.h>

typedef __attribute__((ext_vector_type(16))) _Float16 v16h;
typedef __attribute__((ext_vector_type(8)))  float    v8f;

#define EPSN 1e-5f

static inline int ceilDiv(int a, int b) { return (a + b - 1) / b; }
static inline int gs(long long n) { return (int)((n + 255) / 256); }

// ---------------------------------------------------------------------------
// WMMA implicit-GEMM convolution, 32(M) x 128(N) block tile, K in steps of 32.
// 256 threads = 8 wave32 waves; wave w owns N-columns [16w,16w+16) and computes
// two 16x16 accumulators (M rows 0..15 and 16..31) sharing one B fragment.
//   y[b,o,h,w] = bias[o] + sum_{c,kh,kw} x[b,c,h+kh*dil-pad, w+kw*dil-pad] * wgt[o,c,kh,kw]
// grid: (ceil(HW/128), ceil(O/32), B)
// ---------------------------------------------------------------------------
__global__ __launch_bounds__(256) void conv_wmma(
    const float* __restrict__ x, const float* __restrict__ wgt,
    const float* __restrict__ bias, float* __restrict__ y,
    int C, int H, int W, int O, int KH, int KW, int pad, int dil)
{
  __shared__ _Float16 As[32][32];
  __shared__ _Float16 Bs[32][128];
  const int HW = H * W;
  const int Kt = C * KH * KW;
  const int b = blockIdx.z;
  const int oBase = blockIdx.y * 32;
  const int nBase = blockIdx.x * 128;
  const int tid = threadIdx.x;
  const int lane = tid & 31, wid = tid >> 5;
  v8f acc0 = {}, acc1 = {};
  for (int k0 = 0; k0 < Kt; k0 += 32) {
    // stage A: weight tile 32x32 (row-major (O, C*KH*KW) is already GEMM-A)
    for (int i = tid; i < 32 * 32; i += 256) {
      int m = i >> 5, kk = i & 31;
      int o = oBase + m, k = k0 + kk;
      float v = 0.f;
      if (o < O && k < Kt) v = wgt[(long long)o * Kt + k];
      As[m][kk] = (_Float16)v;
    }
    // stage B: im2col tile 32x128
    for (int i = tid; i < 32 * 128; i += 256) {
      int kk = i >> 7, n = i & 127;
      int k = k0 + kk, pos = nBase + n;
      float v = 0.f;
      if (k < Kt && pos < HW) {
        int c = k / (KH * KW), r = k % (KH * KW);
        int kh = r / KW, kw = r % KW;
        int h = pos / W, w = pos % W;
        int ih = h + kh * dil - pad, iw = w + kw * dil - pad;
        if (ih >= 0 && ih < H && iw >= 0 && iw < W)
          v = x[(((long long)b * C + c) * H + ih) * W + iw];
      }
      Bs[kk][n] = (_Float16)v;
    }
    if (tid == 0 && k0 + 32 < Kt)
      __builtin_prefetch(&wgt[(long long)oBase * Kt + k0 + 32], 0, 0); // global_prefetch_b8
    __syncthreads();
    // fragments per CDNA5 16-bit A(16x32)/B(32x16) wave32 layouts
    v16h a0, a1, bf;
    const int row = lane & 15, hb = lane >> 4;
#pragma unroll
    for (int e = 0; e < 8; ++e) {
      a0[e]     = As[row][hb * 8 + e];           // lanes 0-15: K=0..7 ; 16-31: K=8..15
      a0[e + 8] = As[row][16 + hb * 8 + e];      // lanes 0-15: K=16..23; 16-31: K=24..31
      a1[e]     = As[16 + row][hb * 8 + e];
      a1[e + 8] = As[16 + row][16 + hb * 8 + e];
    }
    const int col = wid * 16 + (lane & 15);
#pragma unroll
    for (int e = 0; e < 16; ++e) bf[e] = Bs[(e < 8 ? hb * 8 + e : 16 + hb * 8 + (e - 8))][col];
    acc0 = __builtin_amdgcn_wmma_f32_16x16x32_f16(false, a0, false, bf, (short)0, acc0,
                                                  false, false);
    acc1 = __builtin_amdgcn_wmma_f32_16x16x32_f16(false, a1, false, bf, (short)0, acc1,
                                                  false, false);
    __syncthreads();
  }
  const int hb = lane >> 4;
#pragma unroll
  for (int j = 0; j < 8; ++j) {
    int m0 = oBase + j + hb * 8;
    int n = nBase + wid * 16 + (lane & 15);
    if (n < HW) {
      if (m0 < O) {
        float v = acc0[j];
        if (bias) v += bias[m0];
        y[((long long)b * O + m0) * HW + n] = v;
      }
      int m1 = m0 + 16;
      if (m1 < O) {
        float v = acc1[j];
        if (bias) v += bias[m1];
        y[((long long)b * O + m1) * HW + n] = v;
      }
    }
  }
}

// ---------------------------------------------------------------------------
// Generic strided batched WMMA GEMM:  Y[b,m,n] = sum_k A[b,m,k] * B[b,k,n]
// A element at A[sAb*b + m*sAm + k*sAk]; B at Bm[sBb*b + k*sBk + n].
// outMode 0: Y[sYb*b + m*N + n]
// outMode 1: transposed-conv scatter: n->(h,w), write [m, 2h+dy, 2w+dx] at 2W pitch
// Same 32x128 tiling as conv_wmma.
// ---------------------------------------------------------------------------
__global__ __launch_bounds__(256) void gemm_wmma(
    const float* __restrict__ A, const float* __restrict__ Bm,
    const float* __restrict__ bias, float* __restrict__ Y,
    int M, int N, int Kt,
    long long sAb, int sAm, int sAk,
    long long sBb, int sBk, long long sYb,
    int outMode, int upH, int upW, int dy, int dx)
{
  __shared__ _Float16 As[32][32];
  __shared__ _Float16 Bs[32][128];
  const int b = blockIdx.z, mBase = blockIdx.y * 32, nBase = blockIdx.x * 128;
  const int tid = threadIdx.x, lane = tid & 31, wid = tid >> 5;
  v8f acc0 = {}, acc1 = {};
  for (int k0 = 0; k0 < Kt; k0 += 32) {
    for (int i = tid; i < 1024; i += 256) {
      int m = i >> 5, kk = i & 31;
      int gm = mBase + m, k = k0 + kk;
      float v = 0.f;
      if (gm < M && k < Kt) v = A[sAb * b + (long long)gm * sAm + (long long)k * sAk];
      As[m][kk] = (_Float16)v;
    }
    for (int i = tid; i < 4096; i += 256) {
      int kk = i >> 7, n = i & 127;
      int k = k0 + kk, gn = nBase + n;
      float v = 0.f;
      if (k < Kt && gn < N) v = Bm[sBb * b + (long long)k * sBk + gn];
      Bs[kk][n] = (_Float16)v;
    }
    if (tid == 0 && k0 + 32 < Kt)
      __builtin_prefetch(&Bm[sBb * b + (long long)(k0 + 32) * sBk + nBase], 0, 0);
    __syncthreads();
    v16h a0, a1, bf;
    const int row = lane & 15, hb = lane >> 4;
#pragma unroll
    for (int e = 0; e < 8; ++e) {
      a0[e]     = As[row][hb * 8 + e];
      a0[e + 8] = As[row][16 + hb * 8 + e];
      a1[e]     = As[16 + row][hb * 8 + e];
      a1[e + 8] = As[16 + row][16 + hb * 8 + e];
    }
    const int col = wid * 16 + (lane & 15);
#pragma unroll
    for (int e = 0; e < 16; ++e) bf[e] = Bs[(e < 8 ? hb * 8 + e : 16 + hb * 8 + (e - 8))][col];
    acc0 = __builtin_amdgcn_wmma_f32_16x16x32_f16(false, a0, false, bf, (short)0, acc0,
                                                  false, false);
    acc1 = __builtin_amdgcn_wmma_f32_16x16x32_f16(false, a1, false, bf, (short)0, acc1,
                                                  false, false);
    __syncthreads();
  }
  const int hb = lane >> 4;
#pragma unroll
  for (int j = 0; j < 8; ++j) {
    int n = nBase + wid * 16 + (lane & 15);
    if (n >= N) continue;
#pragma unroll
    for (int half = 0; half < 2; ++half) {
      int m = mBase + j + hb * 8 + half * 16;
      if (m >= M) continue;
      float v = half ? acc1[j] : acc0[j];
      if (bias) v += bias[m];
      if (outMode == 0) {
        Y[sYb * b + (long long)m * N + n] = v;
      } else {
        int h = n / upW, w = n % upW;
        Y[sYb * b + (long long)m * (4LL * upH * upW) +
          (long long)(2 * h + dy) * (2 * upW) + (2 * w + dx)] = v;
      }
    }
  }
}

// ---------------------------------------------------------------------------
// Tiled transpose with CDNA5 async global->LDS staging.
// out[b, s, q] = in[b, q, s]  (cs x cs per batch, cs multiple of 32)
// grid: (cs/32, cs/32, B), block 256. Each lane async-loads 16B straight into
// LDS (global_load_async_to_lds_b128, ASYNCcnt), then coalesced float4 stores.
// ---------------------------------------------------------------------------
__global__ __launch_bounds__(256) void transpose_qs_k(
    const float* __restrict__ in, float* __restrict__ out, int cs)
{
  __shared__ float tile[32][36];   // 36-float pitch: rows stay 16B aligned
  const int b = blockIdx.z;
  const int s0 = blockIdx.x * 32;  // input column group
  const int q0 = blockIdx.y * 32;  // input row group
  const float* src = in + (long long)b * cs * cs;
  float* dst = out + (long long)b * cs * cs;
  const int t = threadIdx.x;
  const int r = t >> 3;            // 0..31
  const int c4 = (t & 7) << 2;     // 0,4,...,28
  // LDS aperture keeps the byte offset in the low 32 bits of the generic addr.
  unsigned ldsoff = (unsigned)(size_t)&tile[r][c4];
  const float* gp = src + (long long)(q0 + r) * cs + (s0 + c4);
  asm volatile("global_load_async_to_lds_b128 %0, %1, off"
               :: "v"(ldsoff), "v"(gp) : "memory");
  asm volatile("s_wait_asynccnt 0x0" ::: "memory");
  __syncthreads();
  float4 vv;
  vv.x = tile[c4 + 0][r];
  vv.y = tile[c4 + 1][r];
  vv.z = tile[c4 + 2][r];
  vv.w = tile[c4 + 3][r];
  *(float4*)(dst + (long long)(s0 + r) * cs + (q0 + c4)) = vv;
}

// ---------------------------------------------------------------------------
// Normalization / elementwise / reduction kernels
// ---------------------------------------------------------------------------
__global__ __launch_bounds__(256) void bnorm_stats_k(
    const float* __restrict__ x, float* mean, float* rinv, int Bn, int C, int HW)
{
  __shared__ float s1[256], s2[256];
  int c = blockIdx.x;
  float a = 0.f, b = 0.f;
  for (long long i = threadIdx.x; i < (long long)Bn * HW; i += 256) {
    int n = (int)(i / HW), j = (int)(i % HW);
    float v = x[((long long)n * C + c) * HW + j];
    a += v; b += v * v;
  }
  s1[threadIdx.x] = a; s2[threadIdx.x] = b;
  __syncthreads();
  for (int o = 128; o > 0; o >>= 1) {
    if (threadIdx.x < o) { s1[threadIdx.x] += s1[threadIdx.x + o]; s2[threadIdx.x] += s2[threadIdx.x + o]; }
    __syncthreads();
  }
  if (threadIdx.x == 0) {
    float cnt = (float)Bn * (float)HW;
    float m = s1[0] / cnt;
    float v = s2[0] / cnt - m * m;
    mean[c] = m; rinv[c] = rsqrtf(v + EPSN);
  }
}

__global__ __launch_bounds__(256) void bnorm_apply_k(
    const float* __restrict__ x, float* __restrict__ y,
    const float* mean, const float* rinv, const float* g, const float* be,
    int C, int HW, long long total, int doRelu)
{
  long long i = (long long)blockIdx.x * 256 + threadIdx.x;
  if (i >= total) return;
  int c = (int)((i / HW) % C);
  float v = (x[i] - mean[c]) * rinv[c] * g[c] + be[c];
  if (doRelu) v = fmaxf(v, 0.f);
  y[i] = v;
}

__global__ __launch_bounds__(256) void inorm_k(
    const float* __restrict__ x, float* __restrict__ y, int HW, int doRelu)
{
  __shared__ float s1[256], s2[256];
  __shared__ float mS, rS;
  long long base = (long long)blockIdx.x * HW;
  float a = 0.f, b = 0.f;
  for (int i = threadIdx.x; i < HW; i += 256) { float v = x[base + i]; a += v; b += v * v; }
  s1[threadIdx.x] = a; s2[threadIdx.x] = b;
  __syncthreads();
  for (int o = 128; o > 0; o >>= 1) {
    if (threadIdx.x < o) { s1[threadIdx.x] += s1[threadIdx.x + o]; s2[threadIdx.x] += s2[threadIdx.x + o]; }
    __syncthreads();
  }
  if (threadIdx.x == 0) {
    float m = s1[0] / (float)HW;
    float v = s2[0] / (float)HW - m * m;
    mS = m; rS = rsqrtf(v + EPSN);
  }
  __syncthreads();
  for (int i = threadIdx.x; i < HW; i += 256) {
    float v = (x[base + i] - mS) * rS;
    if (doRelu) v = fmaxf(v, 0.f);
    y[base + i] = v;
  }
}

__global__ __launch_bounds__(256) void maxpool_k(
    const float* __restrict__ x, float* __restrict__ y, int C, int H, int W, long long total)
{
  long long i = (long long)blockIdx.x * 256 + threadIdx.x;
  if (i >= total) return;
  int W2 = W / 2, H2 = H / 2;
  int w = (int)(i % W2); long long t = i / W2;
  int h = (int)(t % H2); t /= H2;
  int c = (int)(t % C); int n = (int)(t / C);
  const float* p = x + (((long long)n * C + c) * H + 2 * h) * W + 2 * w;
  y[i] = fmaxf(fmaxf(p[0], p[1]), fmaxf(p[W], p[W + 1]));
}

__global__ __launch_bounds__(256) void add_relu_k(
    const float* __restrict__ a, const float* __restrict__ b, float* __restrict__ y, long long total)
{
  long long i = (long long)blockIdx.x * 256 + threadIdx.x;
  if (i >= total) return;
  y[i] = fmaxf(a[i] + b[i], 0.f);
}

__global__ __launch_bounds__(256) void copy_ch_k(
    const float* __restrict__ src, float* __restrict__ dst,
    int Cs, int HW, int Cd, int coff, long long total)
{
  long long i = (long long)blockIdx.x * 256 + threadIdx.x;
  if (i >= total) return;
  int j = (int)(i % HW); long long t = i / HW;
  int c = (int)(t % Cs); int b = (int)(t / Cs);
  dst[((long long)b * Cd + coff + c) * HW + j] = src[i];
}

__global__ __launch_bounds__(256) void mask_mul_k(
    const float* __restrict__ corr, const float* __restrict__ m, float* __restrict__ cm,
    int cs, int inv, long long total)
{
  long long i = (long long)blockIdx.x * 256 + threadIdx.x;
  if (i >= total) return;
  long long t = i / cs;
  int q = (int)(t % cs); int b = (int)(t / cs);
  float mv = m[(long long)b * cs + q];
  if (inv) mv = 1.f - mv;
  cm[i] = corr[i] * mv;
}

__global__ __launch_bounds__(256) void proto_k(
    const float* __restrict__ sup, const float* __restrict__ m, float* __restrict__ proto,
    int C, int HW)
{
  __shared__ float s1[256], s2[256];
  int bc = blockIdx.x; int c = bc % C; int b = bc / C;
  float a = 0.f, mm = 0.f;
  for (int i = threadIdx.x; i < HW; i += 256) {
    float mv = m[(long long)b * HW + i];
    a += sup[((long long)b * C + c) * HW + i] * mv;
    mm += mv;
  }
  s1[threadIdx.x] = a; s2[threadIdx.x] = mm;
  __syncthreads();
  for (int o = 128; o > 0; o >>= 1) {
    if (threadIdx.x < o) { s1[threadIdx.x] += s1[threadIdx.x + o]; s2[threadIdx.x] += s2[threadIdx.x + o]; }
    __syncthreads();
  }
  if (threadIdx.x == 0) proto[bc] = s1[0] / s2[0];
}

__global__ __launch_bounds__(256) void cos_k(
    const float* __restrict__ que, const float* __restrict__ proto, float* __restrict__ out,
    int C, int HW, long long total)
{
  long long i = (long long)blockIdx.x * 256 + threadIdx.x;
  if (i >= total) return;
  int b = (int)(i / HW); int j = (int)(i % HW);
  float num = 0.f, q2 = 0.f, p2 = 0.f;
  for (int c = 0; c < C; ++c) {
    float qv = que[((long long)b * C + c) * HW + j];
    float pv = proto[b * C + c];
    num += qv * pv; q2 += qv * qv; p2 += pv * pv;
  }
  float den = fmaxf(sqrtf(q2) * sqrtf(p2), 1e-8f);
  out[i] = num / den;
}

__global__ __launch_bounds__(256) void score_k(
    const float* __restrict__ cm, const float* __restrict__ rank, const float* __restrict__ m,
    float* __restrict__ score, int cs, int inv)
{
  __shared__ float s1[256];
  int bq = blockIdx.x;
  long long base = (long long)bq * cs;
  float v = -__builtin_inff();
  for (int i = threadIdx.x; i < cs; i += 256) v = fmaxf(v, cm[base + i]);
  s1[threadIdx.x] = v;
  __syncthreads();
  for (int o = 128; o > 0; o >>= 1) {
    if (threadIdx.x < o) s1[threadIdx.x] = fmaxf(s1[threadIdx.x], s1[threadIdx.x + o]);
    __syncthreads();
  }
  if (threadIdx.x == 0) {
    int b = bq / cs; int q = bq % cs;
    float mv = m[(long long)b * cs + q];
    if (inv) mv = 1.f - mv;
    score[bq] = s1[0] + fmaxf(rank[bq], 0.f) * mv;
  }
}

__global__ __launch_bounds__(256) void topk_k(
    const float* __restrict__ score, int* __restrict__ idx, int cs, int Ksel)
{
  __shared__ float sv[4096];
  __shared__ float rv[256];
  __shared__ int ri[256];
  int b = blockIdx.x;
  for (int i = threadIdx.x; i < cs; i += 256) sv[i] = score[(long long)b * cs + i];
  __syncthreads();
  for (int k = 0; k < Ksel; ++k) {
    float best = -__builtin_inff(); int bi = 0;
    for (int i = threadIdx.x; i < cs; i += 256)
      if (sv[i] > best) { best = sv[i]; bi = i; }
    rv[threadIdx.x] = best; ri[threadIdx.x] = bi;
    __syncthreads();
    for (int o = 128; o > 0; o >>= 1) {
      if (threadIdx.x < o && rv[threadIdx.x + o] > rv[threadIdx.x]) {
        rv[threadIdx.x] = rv[threadIdx.x + o]; ri[threadIdx.x] = ri[threadIdx.x + o];
      }
      __syncthreads();
    }
    if (threadIdx.x == 0) { idx[b * Ksel + k] = ri[0]; sv[ri[0]] = -__builtin_inff(); }
    __syncthreads();
  }
}

__global__ __launch_bounds__(256) void gather_k(
    const float* __restrict__ cm, const int* __restrict__ idx, float* __restrict__ sel,
    int Ksel, int cs, long long total)
{
  long long i = (long long)blockIdx.x * 256 + threadIdx.x;
  if (i >= total) return;
  int s = (int)(i % cs); long long t = i / cs;
  int k = (int)(t % Ksel); int b = (int)(t / Ksel);
  sel[i] = cm[((long long)b * cs + idx[b * Ksel + k]) * cs + s];
}

__global__ __launch_bounds__(256) void sigmoid_k(
    const float* __restrict__ x, float* __restrict__ y, long long total)
{
  long long i = (long long)blockIdx.x * 256 + threadIdx.x;
  if (i >= total) return;
  y[i] = 1.f / (1.f + expf(-x[i]));
}

// ---------------------------------------------------------------------------
// Host-side orchestration
// ---------------------------------------------------------------------------
namespace {

struct Par { void* const* din; int i; const float* f() { return (const float*)din[i++]; } };
struct DCP { const float *w1, *b1, *g1, *be1, *w2, *b2, *g2, *be2; };
struct UPP { const float *w, *b; DCP dc; };
struct UNetP { DCP inc, d1, d2, d3, d4; UPP u1, u2, u3, u4; };
struct RankP { const float *wa, *ba, *c1, *c2, *c3, *wb, *bb; };

DCP getDCP(Par& p) {
  DCP d; d.w1 = p.f(); d.b1 = p.f(); d.g1 = p.f(); d.be1 = p.f();
  d.w2 = p.f(); d.b2 = p.f(); d.g2 = p.f(); d.be2 = p.f(); return d;
}
UPP getUPP(Par& p) { UPP u; u.w = p.f(); u.b = p.f(); u.dc = getDCP(p); return u; }

struct WS {
  char* base; size_t off;
  float* f(size_t n) {
    float* r = (float*)(base + off);
    off += ((n * sizeof(float) + 255) & ~(size_t)255);
    return r;
  }
  int* i32(size_t n) { return (int*)f(n); }
};

void launch_conv(hipStream_t st, const float* x, const float* w, const float* bias,
                 float* y, int Bn, int C, int H, int W, int O, int KH, int KW, int pad, int dil)
{
  dim3 g(ceilDiv(H * W, 128), ceilDiv(O, 32), Bn);
  conv_wmma<<<g, dim3(256), 0, st>>>(x, w, bias, y, C, H, W, O, KH, KW, pad, dil);
}

void launch_bnorm(hipStream_t st, float* x, const float* g, const float* be,
                  float* mean, float* rinv, int Bn, int C, int HW, int relu)
{
  bnorm_stats_k<<<C, 256, 0, st>>>(x, mean, rinv, Bn, C, HW);
  long long total = (long long)Bn * C * HW;
  bnorm_apply_k<<<gs(total), 256, 0, st>>>(x, x, mean, rinv, g, be, C, HW, total, relu);
}

void dconv_h(hipStream_t st, WS ws, const float* x, float* out,
             const DCP& p, int Bn, int Ci, int Co, int H, int W)
{
  float* tmp  = ws.f((size_t)Bn * Co * H * W);
  float* mean = ws.f(Co);
  float* rinv = ws.f(Co);
  launch_conv(st, x, p.w1, p.b1, tmp, Bn, Ci, H, W, Co, 3, 3, 1, 1);
  launch_bnorm(st, tmp, p.g1, p.be1, mean, rinv, Bn, Co, H * W, 1);
  launch_conv(st, tmp, p.w2, p.b2, out, Bn, Co, H, W, Co, 3, 3, 1, 1);
  launch_bnorm(st, out, p.g2, p.be2, mean, rinv, Bn, Co, H * W, 1);
}

void launch_upconv(hipStream_t st, const float* x, const float* w, const float* bias,
                   float* y, int Bn, int C, int H, int W, int O)
{
  int HW = H * W;
  for (int dy = 0; dy < 2; ++dy)
    for (int dx = 0; dx < 2; ++dx) {
      dim3 g(ceilDiv(HW, 128), ceilDiv(O, 32), Bn);
      // A[m=o,k=c] = w[c,o,dy,dx] : strideAm = 4, strideAk = O*4, base offset dy*2+dx
      gemm_wmma<<<g, dim3(256), 0, st>>>(w + (dy * 2 + dx), x, bias, y,
          O, HW, C,
          0LL, 4, O * 4,
          (long long)C * HW, HW,
          (long long)O * 4 * HW,
          1, H, W, dy, dx);
    }
}

void up_h(hipStream_t st, WS ws, const float* a, const float* skip, float* out,
          const UPP& u, int Bn, int Ca, int H, int W)
{
  int O = Ca / 2, H2 = 2 * H, W2 = 2 * W, HW2 = H2 * W2;
  float* uo  = ws.f((size_t)Bn * O * HW2);
  float* cat = ws.f((size_t)Bn * Ca * HW2);
  launch_upconv(st, a, u.w, u.b, uo, Bn, Ca, H, W, O);
  long long tc = (long long)Bn * O * HW2;
  copy_ch_k<<<gs(tc), 256, 0, st>>>(skip, cat, O, HW2, Ca, 0, tc);
  copy_ch_k<<<gs(tc), 256, 0, st>>>(uo,   cat, O, HW2, Ca, O, tc);
  dconv_h(st, ws, cat, out, u.dc, Bn, Ca, O, H2, W2);
}

void launch_maxpool(hipStream_t st, const float* x, float* y, int Bn, int C, int H, int W)
{
  long long total = (long long)Bn * C * (H / 2) * (W / 2);
  maxpool_k<<<gs(total), 256, 0, st>>>(x, y, C, H, W, total);
}

void unet_fwd(hipStream_t st, WS ws, const float* img, float* out, const UNetP& U, int Bn)
{
  const int bd = 64;
  float* x1 = ws.f((size_t)Bn * bd * 4096);
  float* x2 = ws.f((size_t)Bn * 2 * bd * 1024);
  float* x3 = ws.f((size_t)Bn * 4 * bd * 256);
  float* x4 = ws.f((size_t)Bn * 8 * bd * 64);
  float* x5 = ws.f((size_t)Bn * 16 * bd * 16);
  float* p1 = ws.f((size_t)Bn * bd * 1024);
  float* p2 = ws.f((size_t)Bn * 2 * bd * 256);
  float* p3 = ws.f((size_t)Bn * 4 * bd * 64);
  float* p4 = ws.f((size_t)Bn * 8 * bd * 16);
  float* o4 = ws.f((size_t)Bn * 8 * bd * 64);
  float* o3 = ws.f((size_t)Bn * 4 * bd * 256);
  float* o2 = ws.f((size_t)Bn * 2 * bd * 1024);

  dconv_h(st, ws, img, x1, U.inc, Bn, 1, bd, 64, 64);
  launch_maxpool(st, x1, p1, Bn, bd, 64, 64);
  dconv_h(st, ws, p1, x2, U.d1, Bn, bd, 2 * bd, 32, 32);
  launch_maxpool(st, x2, p2, Bn, 2 * bd, 32, 32);
  dconv_h(st, ws, p2, x3, U.d2, Bn, 2 * bd, 4 * bd, 16, 16);
  launch_maxpool(st, x3, p3, Bn, 4 * bd, 16, 16);
  dconv_h(st, ws, p3, x4, U.d3, Bn, 4 * bd, 8 * bd, 8, 8);
  launch_maxpool(st, x4, p4, Bn, 8 * bd, 8, 8);
  dconv_h(st, ws, p4, x5, U.d4, Bn, 8 * bd, 16 * bd, 4, 4);

  up_h(st, ws, x5, x4, o4, U.u1, Bn, 16 * bd, 4, 4);
  up_h(st, ws, o4, x3, o3, U.u2, Bn, 8 * bd, 8, 8);
  up_h(st, ws, o3, x2, o2, U.u3, Bn, 4 * bd, 16, 16);
  up_h(st, ws, o2, x1, out, U.u4, Bn, 2 * bd, 32, 32);
}

// ml2: conv(relu(inorm(x)), w, b, pad=1).  x is normalized in place.
void ml2_h(hipStream_t st, float* x, float* y, const float* w, const float* b,
           int Bn, int Ci, int Co, int H, int W)
{
  inorm_k<<<Bn * Ci, 256, 0, st>>>(x, x, H * W, 1);
  launch_conv(st, x, w, b, y, Bn, Ci, H, W, Co, 3, 3, 1, 1);
}

void resblock_h(hipStream_t st, WS ws, const float* x, float* out,
                const float* c1, const float* c2, const float* c3,
                int Bn, int Cin, int planes, int H, int W)
{
  int HW = H * W;
  float* t1 = ws.f((size_t)Bn * planes * HW);
  float* t2 = ws.f((size_t)Bn * planes * HW);
  float* t3 = ws.f((size_t)Bn * Cin * HW);
  launch_conv(st, x, c1, nullptr, t1, Bn, Cin, H, W, planes, 1, 1, 0, 1);
  inorm_k<<<Bn * planes, 256, 0, st>>>(t1, t1, HW, 1);
  launch_conv(st, t1, c2, nullptr, t2, Bn, planes, H, W, planes, 3, 3, 1, 1);
  inorm_k<<<Bn * planes, 256, 0, st>>>(t2, t2, HW, 1);
  launch_conv(st, t2, c3, nullptr, t3, Bn, planes, H, W, Cin, 1, 1, 0, 1);
  inorm_k<<<Bn * Cin, 256, 0, st>>>(t3, t3, HW, 0);
  long long total = (long long)Bn * Cin * HW;
  add_relu_k<<<gs(total), 256, 0, st>>>(t3, x, out, total);
}

void branch_h(hipStream_t st, WS ws, const float* corr, const float* mask, int inv,
              float* outp, const RankP& rk,
              const float* p1w, const float* p1b,
              const float* r1c1, const float* r1c2, const float* r1c3,
              const float* p2w, const float* p2b, int Bn)
{
  const int cs = 4096;
  long long tot = (long long)Bn * cs * cs;
  float* cm = ws.f((size_t)tot);
  float* T  = ws.f((size_t)tot);
  mask_mul_k<<<gs(tot), 256, 0, st>>>(corr, mask, cm, cs, inv, tot);
  {
    dim3 g(cs / 32, cs / 32, Bn);
    transpose_qs_k<<<g, dim3(256), 0, st>>>(cm, T, cs);
  }

  // ranking: ml2(wa) -> resblock(planes 32) -> ml2(wb)
  float* r1 = ws.f((size_t)Bn * 128 * cs);
  float* r2 = ws.f((size_t)Bn * 128 * cs);
  float* rout = ws.f((size_t)Bn * cs);
  ml2_h(st, T, r1, rk.wa, rk.ba, Bn, 4096, 128, 64, 64);
  resblock_h(st, ws, r1, r2, rk.c1, rk.c2, rk.c3, Bn, 128, 32, 64, 64);
  ml2_h(st, r2, rout, rk.wb, rk.bb, Bn, 128, 1, 64, 64);

  float* score = ws.f((size_t)Bn * cs);
  score_k<<<Bn * cs, 256, 0, st>>>(cm, rout, mask, score, cs, inv);
  int* idx = ws.i32((size_t)Bn * 128);
  topk_k<<<Bn, 256, 0, st>>>(score, idx, cs, 128);
  float* sel = ws.f((size_t)Bn * 128 * cs);
  long long tg = (long long)Bn * 128 * cs;
  gather_k<<<gs(tg), 256, 0, st>>>(cm, idx, sel, 128, cs, tg);

  float* h1 = ws.f((size_t)Bn * 128 * cs);
  ml2_h(st, sel, h1, p1w, p1b, Bn, 128, 128, 64, 64);
  float* h2 = ws.f((size_t)Bn * 128 * cs);
  resblock_h(st, ws, h1, h2, r1c1, r1c2, r1c3, Bn, 128, 128, 64, 64);
  ml2_h(st, h2, outp, p2w, p2b, Bn, 128, 64, 64, 64);
}

} // namespace

extern "C" void kernel_launch(void* const* d_in, const int* in_sizes, int n_in,
                              void* d_out, int out_size, void* d_ws, size_t ws_size,
                              hipStream_t stream)
{
  (void)in_sizes; (void)n_in; (void)out_size; (void)ws_size;
  Par p{d_in, 0};
  const float* query   = p.f();
  const float* support = p.f();
  const float* mask    = p.f();

  UNetP U;
  U.inc = getDCP(p); U.d1 = getDCP(p); U.d2 = getDCP(p); U.d3 = getDCP(p); U.d4 = getDCP(p);
  U.u1 = getUPP(p); U.u2 = getUPP(p); U.u3 = getUPP(p); U.u4 = getUPP(p);
  RankP rk;
  rk.wa = p.f(); rk.ba = p.f(); rk.c1 = p.f(); rk.c2 = p.f(); rk.c3 = p.f();
  rk.wb = p.f(); rk.bb = p.f();
  const float* p1w = p.f(); const float* p1b = p.f();
  const float* r1c1 = p.f(); const float* r1c2 = p.f(); const float* r1c3 = p.f();
  const float* p2w = p.f(); const float* p2b = p.f();
  DCP segw = getDCP(p);
  const float* predw = p.f(); const float* predb = p.f();

  WS ws{(char*)d_ws, 0};
  const int Bn = 2, C = 64, HW = 4096, cs = 4096;

  float* sup4 = ws.f((size_t)Bn * C * HW);
  float* que4 = ws.f((size_t)Bn * C * HW);
  unet_fwd(stream, ws, support, sup4, U, Bn);
  unet_fwd(stream, ws, query,   que4, U, Bn);

  float* proto = ws.f((size_t)Bn * C);
  proto_k<<<Bn * C, 256, 0, stream>>>(sup4, mask, proto, C, HW);
  float* cosb = ws.f((size_t)Bn * HW);
  long long tcos = (long long)Bn * HW;
  cos_k<<<gs(tcos), 256, 0, stream>>>(que4, proto, cosb, C, HW, tcos);

  // correlation: corr[b,q,s] = sum_c que[b,c,q] * sup[b,c,s]  (A^T * B via strides)
  float* corr = ws.f((size_t)Bn * cs * cs);
  {
    dim3 g(ceilDiv(cs, 128), ceilDiv(cs, 32), Bn);
    gemm_wmma<<<g, dim3(256), 0, stream>>>(que4, sup4, nullptr, corr,
        cs, cs, C,
        (long long)C * cs, 1, cs,
        (long long)C * cs, cs,
        (long long)cs * cs,
        0, 0, 0, 0, 0);
  }

  float* fs = ws.f((size_t)Bn * 64 * HW);
  float* bs = ws.f((size_t)Bn * 64 * HW);
  branch_h(stream, ws, corr, mask, 0, fs, rk, p1w, p1b, r1c1, r1c2, r1c3, p2w, p2b, Bn);
  branch_h(stream, ws, corr, mask, 1, bs, rk, p1w, p1b, r1c1, r1c2, r1c3, p2w, p2b, Bn);

  // feat = concat([que4(64), cos(1), fs(64), bs(64)]) -> 193 channels
  const int Cf = 193;
  float* feat = ws.f((size_t)Bn * Cf * HW);
  long long t64 = (long long)Bn * 64 * HW, t1 = (long long)Bn * HW;
  copy_ch_k<<<gs(t64), 256, 0, stream>>>(que4, feat, 64, HW, Cf, 0, t64);
  copy_ch_k<<<gs(t1),  256, 0, stream>>>(cosb, feat, 1,  HW, Cf, 64, t1);
  copy_ch_k<<<gs(t64), 256, 0, stream>>>(fs,   feat, 64, HW, Cf, 65, t64);
  copy_ch_k<<<gs(t64), 256, 0, stream>>>(bs,   feat, 64, HW, Cf, 129, t64);

  float* segx = ws.f((size_t)Bn * 64 * HW);
  dconv_h(stream, ws, feat, segx, segw, Bn, Cf, 64, 64, 64);

  float* logits = ws.f((size_t)Bn * HW);
  launch_conv(stream, segx, predw, predb, logits, Bn, 64, 64, 64, 1, 1, 1, 0, 1);
  sigmoid_k<<<gs((long long)Bn * HW), 256, 0, stream>>>(logits, (float*)d_out,
                                                        (long long)Bn * HW);
}
